// AttentionWithRotary_14061722927478
// MI455X (gfx1250) — compile-verified, matrix operand order checked by reading, not run on hardware
//
#include <hip/hip_runtime.h>
#include <hip/hip_bf16.h>

// ---------------------------------------------------------------------------
// AttentionWithRotary for MI455X (gfx1250, wave32, WMMA bf16, async-to-LDS)
// D = 2048, H = 16, hd = 128, B = 2, S = 2048
// Pipeline: QKV proj (WMMA) -> RoPE -> streaming QK^T*V per head (WMMA)
//           -> output proj (WMMA, fp32 out)
// ---------------------------------------------------------------------------

#define DIM      2048
#define N_HEADS  16
#define HEAD_DIM 128
#define BATCH    2
#define SEQ      2048
#define MTOT     (BATCH * SEQ)          // 4096 rows
#define SCALE_P  0.08838834764831845f   // 1/sqrt(128)

typedef __attribute__((ext_vector_type(16))) __bf16 v16bf;
typedef __attribute__((ext_vector_type(8)))  float  v8f;

struct alignas(8) bf4 { __bf16 v[4]; };

// LDS byte address (low 32 bits of generic shared pointer == LDS offset).
__device__ inline uint32_t lds_addr(const void* p) {
    return (uint32_t)(size_t)p;
}

// Async copy 16B per lane: global -> LDS, tracked on ASYNCcnt (no VGPR data).
__device__ inline void async_copy_b128(uint32_t lds, const void* gptr) {
    asm volatile("global_load_async_to_lds_b128 %0, %1, off"
                 :: "v"(lds), "v"(gptr) : "memory");
}
__device__ inline void wait_asynccnt0() {
    asm volatile("s_wait_asynccnt 0x0" ::: "memory");
}

// Load one 16x32 bf16 A/B fragment (ISA 7.12.2 layout) from a row-major tile.
// `p` points at this lane's row start (row = base + (lane&15)), k-offset applied.
// lanes 0-15 read K[0..7] and K[16..23]; lanes 16-31 read K[8..15] and K[24..31].
__device__ inline v16bf ld_frag(const __bf16* p, int lane) {
    const uint4* q = (const uint4*)p;
    int hi = (lane >> 4) & 1;
    union { uint4 u[2]; v16bf v; } r;
    r.u[0] = q[hi];
    r.u[1] = q[2 + hi];
    return r.v;
}

__device__ inline v8f wmma_bf16(v16bf a, v16bf b, v8f c) {
    return __builtin_amdgcn_wmma_f32_16x16x32_bf16(
        false, a, false, b, (short)0, c, false, false);
}

// ---------------------------------------------------------------------------
// Tiled GEMM:  out[M,N] = A[M,K] @ W[N,K]^T + bias[N]
// 128x128 tile / block of 256 threads (8 waves); each wave owns 32x64.
// KTILE=64: 16 WMMAs per wave between barriers. A fp32 (convert) or bf16
// (async-to-LDS DMA); W always fp32; out bf16 or fp32.
// ---------------------------------------------------------------------------
template<bool A_IS_BF16, bool OUT_FLOAT>
__global__ __launch_bounds__(256)
void gemm_bias_kernel(const void* __restrict__ A_, const float* __restrict__ W,
                      const float* __restrict__ bias, void* __restrict__ out,
                      int M, int N, int K) {
    __shared__ __attribute__((aligned(16))) __bf16 As[128][72]; // 64 valid + pad
    __shared__ __attribute__((aligned(16))) __bf16 Bs[128][72];

    const int t    = threadIdx.x;
    const int lane = t & 31;
    const int w    = t >> 5;
    const int wm   = w & 3;          // 4 row-groups of 32
    const int wn   = w >> 2;         // 2 col-groups of 64
    const int mBase = blockIdx.y * 128;
    const int nBase = blockIdx.x * 128;

    const int lrow = t >> 1;         // 0..127, tile row this thread fills
    const int lch  = (t & 1) * 32;   // k half (32 elements)

    v8f acc[2][4];
    for (int mi = 0; mi < 2; ++mi)
        for (int ni = 0; ni < 4; ++ni)
            acc[mi][ni] = (v8f){0.f,0.f,0.f,0.f,0.f,0.f,0.f,0.f};

    for (int k0 = 0; k0 < K; k0 += 64) {
        // ---- stage A tile ----
        if (A_IS_BF16) {
            const __bf16* ag = (const __bf16*)A_ + (size_t)(mBase + lrow) * K + k0 + lch;
            const uint32_t la = lds_addr(&As[lrow][lch]);
            #pragma unroll
            for (int j = 0; j < 4; ++j)
                async_copy_b128(la + j * 16, ag + j * 8);
            __builtin_prefetch(ag + 64, 0, 1);
        } else {
            const float* ag = (const float*)A_ + (size_t)(mBase + lrow) * K + k0 + lch;
            #pragma unroll
            for (int j = 0; j < 8; ++j) {
                float4 f = ((const float4*)ag)[j];
                bf4 pk; pk.v[0]=(__bf16)f.x; pk.v[1]=(__bf16)f.y;
                        pk.v[2]=(__bf16)f.z; pk.v[3]=(__bf16)f.w;
                *(bf4*)&As[lrow][lch + 4*j] = pk;
            }
            __builtin_prefetch(ag + 64, 0, 1);
        }
        // ---- stage W tile (rows nBase..nBase+127 of W), fp32 -> bf16 ----
        {
            const float* bg = W + (size_t)(nBase + lrow) * K + k0 + lch;
            #pragma unroll
            for (int j = 0; j < 8; ++j) {
                float4 f = ((const float4*)bg)[j];
                bf4 pk; pk.v[0]=(__bf16)f.x; pk.v[1]=(__bf16)f.y;
                        pk.v[2]=(__bf16)f.z; pk.v[3]=(__bf16)f.w;
                *(bf4*)&Bs[lrow][lch + 4*j] = pk;
            }
            __builtin_prefetch(bg + 64, 0, 1);
        }
        if (A_IS_BF16) wait_asynccnt0();
        __syncthreads();

        #pragma unroll
        for (int kk = 0; kk < 2; ++kk) {
            v16bf af[2], bfr[4];
            #pragma unroll
            for (int mi = 0; mi < 2; ++mi)
                af[mi] = ld_frag(&As[wm*32 + mi*16 + (lane & 15)][kk*32], lane);
            #pragma unroll
            for (int ni = 0; ni < 4; ++ni)
                bfr[ni] = ld_frag(&Bs[wn*64 + ni*16 + (lane & 15)][kk*32], lane);
            #pragma unroll
            for (int mi = 0; mi < 2; ++mi)
                #pragma unroll
                for (int ni = 0; ni < 4; ++ni)
                    acc[mi][ni] = wmma_bf16(af[mi], bfr[ni], acc[mi][ni]);
        }
        __syncthreads();
    }

    // ---- epilogue: bias + store (C layout: N on lanes, M across 8 VGPRs) ----
    #pragma unroll
    for (int ni = 0; ni < 4; ++ni) {
        const int n  = nBase + wn*64 + ni*16 + (lane & 15);
        const float bv = bias[n];
        #pragma unroll
        for (int mi = 0; mi < 2; ++mi) {
            const int m0 = mBase + wm*32 + mi*16 + ((lane >> 4) << 3);
            #pragma unroll
            for (int i = 0; i < 8; ++i) {
                float val = acc[mi][ni][i] + bv;
                if (OUT_FLOAT)
                    ((float*)out)[(size_t)(m0 + i) * N + n] = val;
                else
                    ((__bf16*)out)[(size_t)(m0 + i) * N + n] = (__bf16)val;
            }
        }
    }
}

// ---------------------------------------------------------------------------
// RoPE (interleaved pairs) applied in-place to bf16 Q and K.
// ---------------------------------------------------------------------------
__global__ __launch_bounds__(256)
void rope_kernel(__bf16* __restrict__ q, __bf16* __restrict__ k,
                 const float* __restrict__ fcos, const float* __restrict__ fsin) {
    int id = blockIdx.x * blockDim.x + threadIdx.x;    // pair index
    const int PAIRS_PER_ROW = DIM / 2;                 // 1024
    int r  = id >> 10;                                 // row in [0, B*S)
    int c  = id & (PAIRS_PER_ROW - 1);
    int pp = c & (HEAD_DIM/2 - 1);                     // pair within head
    int s  = r & (SEQ - 1);                            // sequence position
    float cv = fcos[s * (HEAD_DIM/2) + pp];
    float sv = fsin[s * (HEAD_DIM/2) + pp];
    size_t off = (size_t)r * DIM + 2 * c;              // h*128 + 2*pp == 2*c

    float q0 = (float)q[off], q1 = (float)q[off + 1];
    q[off]     = (__bf16)(q0 * cv - q1 * sv);
    q[off + 1] = (__bf16)(q0 * sv + q1 * cv);
    float k0 = (float)k[off], k1 = (float)k[off + 1];
    k[off]     = (__bf16)(k0 * cv - k1 * sv);
    k[off + 1] = (__bf16)(k0 * sv + k1 * cv);
}

// ---------------------------------------------------------------------------
// Streaming attention (no softmax in reference): O = (Q K^T * scale) V
// One block per (b, h, 64-row q tile); 4 waves, each owns 16 q rows.
// K blocks DMA'd to LDS via async-to-LDS; V transposed at fill (packed b32);
// P (64x64) re-staged via LDS as bf16 to turn C-fragments into A-fragments.
// ---------------------------------------------------------------------------
__global__ __launch_bounds__(128)
void attn_kernel(const __bf16* __restrict__ q, const __bf16* __restrict__ k,
                 const __bf16* __restrict__ v, __bf16* __restrict__ attn) {
    __shared__ __attribute__((aligned(16))) __bf16 Ks[64][136];   // 64 x 128 (+pad)
    __shared__ __attribute__((aligned(16))) __bf16 Vt[128][72];   // transposed V
    __shared__ __attribute__((aligned(16))) __bf16 Ps[64][72];    // P tile bf16

    const int t    = threadIdx.x;
    const int lane = t & 31;
    const int w    = t >> 5;               // wave id: q-row group
    const int qt   = blockIdx.x;           // q tile (64 rows)
    const int h    = blockIdx.y;
    const int b    = blockIdx.z;

    // Q fragments for this wave: 16 rows x 128 k, loaded once from global.
    const __bf16* qbase = q + (size_t)(b*SEQ + qt*64 + w*16 + (lane & 15)) * DIM
                            + h * HEAD_DIM;
    v16bf Qf[4];
    #pragma unroll
    for (int kk = 0; kk < 4; ++kk)
        Qf[kk] = ld_frag(qbase + kk * 32, lane);

    v8f O[8];
    #pragma unroll
    for (int d = 0; d < 8; ++d)
        O[d] = (v8f){0.f,0.f,0.f,0.f,0.f,0.f,0.f,0.f};

    const int lr = t >> 1;                 // 0..63  K-tile row this thread fills
    const int lc = (t & 1) * 64;           // K-tile column half
    const int vr = (t >> 2) * 2;           // V rows (pair) this thread owns
    const int vc = (t & 3) * 32;           // V column chunk (32 cols)

    for (int kb = 0; kb < SEQ / 64; ++kb) {
        const __bf16* kg = k + (size_t)(b*SEQ + kb*64) * DIM + h * HEAD_DIM;
        const __bf16* vg = v + (size_t)(b*SEQ + kb*64) * DIM + h * HEAD_DIM;

        // K tile: async DMA global->LDS (16B/lane, ASYNCcnt).
        {
            const __bf16* src = kg + (size_t)lr * DIM + lc;
            const uint32_t la = lds_addr(&Ks[lr][lc]);
            #pragma unroll
            for (int j = 0; j < 8; ++j)
                async_copy_b128(la + j * 16, src + j * 8);
            __builtin_prefetch(src + 64 * DIM, 0, 1);
        }
        // V tile: transpose 2 rows x 32 cols per thread, packed b32 stores.
        {
            union { uint4 u[4]; __bf16 e[32]; } r0, r1;
            #pragma unroll
            for (int i = 0; i < 4; ++i) {
                r0.u[i] = *(const uint4*)(vg + (size_t)vr       * DIM + vc + i*8);
                r1.u[i] = *(const uint4*)(vg + (size_t)(vr + 1) * DIM + vc + i*8);
            }
            __builtin_prefetch(vg + (size_t)vr * DIM + vc + 64 * DIM, 0, 1);
            #pragma unroll
            for (int d = 0; d < 32; ++d) {
                union { __bf16 h2[2]; uint32_t u; } pk;
                pk.h2[0] = r0.e[d];
                pk.h2[1] = r1.e[d];
                *(uint32_t*)&Vt[vc + d][vr] = pk.u;   // vr even -> 4B aligned
            }
        }
        wait_asynccnt0();
        __syncthreads();

        // P = Q K^T * scale  (this wave: rows w*16.., all 64 cols)
        #pragma unroll
        for (int ni = 0; ni < 4; ++ni) {
            v8f p = (v8f){0.f,0.f,0.f,0.f,0.f,0.f,0.f,0.f};
            #pragma unroll
            for (int kk = 0; kk < 4; ++kk)
                p = wmma_bf16(Qf[kk],
                              ld_frag(&Ks[ni*16 + (lane & 15)][kk*32], lane), p);
            const int col   = ni*16 + (lane & 15);
            const int rbase = w*16 + ((lane >> 4) << 3);
            #pragma unroll
            for (int i = 0; i < 8; ++i)
                Ps[rbase + i][col] = (__bf16)(p[i] * SCALE_P);
        }
        __syncthreads();

        // O += P V   (A-frags from Ps, B-frags from Vt)
        #pragma unroll
        for (int jf = 0; jf < 2; ++jf) {
            v16bf Ap = ld_frag(&Ps[w*16 + (lane & 15)][jf*32], lane);
            #pragma unroll
            for (int d = 0; d < 8; ++d) {
                v16bf Bv = ld_frag(&Vt[d*16 + (lane & 15)][jf*32], lane);
                O[d] = wmma_bf16(Ap, Bv, O[d]);
            }
        }
        __syncthreads();
    }

    // Write O (bf16) into attn intermediate, (b, s, h*128 + d) layout.
    __bf16* ab = attn + (size_t)(b*SEQ + qt*64 + w*16 + ((lane >> 4) << 3)) * DIM
                      + h * HEAD_DIM;
    #pragma unroll
    for (int d = 0; d < 8; ++d)
        #pragma unroll
        for (int i = 0; i < 8; ++i)
            ab[(size_t)i * DIM + d*16 + (lane & 15)] = (__bf16)(O[d][i]);
}

// ---------------------------------------------------------------------------
extern "C" void kernel_launch(void* const* d_in, const int* in_sizes, int n_in,
                              void* d_out, int out_size, void* d_ws, size_t ws_size,
                              hipStream_t stream) {
    const float* x    = (const float*)d_in[0];
    const float* fcos = (const float*)d_in[1];
    const float* fsin = (const float*)d_in[2];
    const float* wq_w = (const float*)d_in[3];
    const float* wq_b = (const float*)d_in[4];
    const float* wk_w = (const float*)d_in[5];
    const float* wk_b = (const float*)d_in[6];
    const float* wv_w = (const float*)d_in[7];
    const float* wv_b = (const float*)d_in[8];
    const float* wo_w = (const float*)d_in[9];
    const float* wo_b = (const float*)d_in[10];

    const size_t MAT = (size_t)MTOT * DIM;       // 8.4M elements
    __bf16* qb = (__bf16*)d_ws;
    __bf16* kb = qb + MAT;
    __bf16* vb = kb + MAT;
    __bf16* ab = vb + MAT;                       // attn intermediate

    dim3 gblk(256);
    dim3 ggrid(DIM / 128, MTOT / 128);           // (16, 32)

    // Q/K/V projections (fp32 A, fp32 W -> bf16 out)
    gemm_bias_kernel<false, false><<<ggrid, gblk, 0, stream>>>(x, wq_w, wq_b, qb, MTOT, DIM, DIM);
    gemm_bias_kernel<false, false><<<ggrid, gblk, 0, stream>>>(x, wk_w, wk_b, kb, MTOT, DIM, DIM);
    gemm_bias_kernel<false, false><<<ggrid, gblk, 0, stream>>>(x, wv_w, wv_b, vb, MTOT, DIM, DIM);

    // RoPE in-place on Q, K
    const int pairs = MTOT * (DIM / 2);
    rope_kernel<<<pairs / 256, 256, 0, stream>>>(qb, kb, fcos, fsin);

    // Attention: O = (QK^T/sqrt(hd)) V, streamed per (b, h, q-tile)
    dim3 agrid(SEQ / 64, N_HEADS, BATCH);        // (32, 16, 2)
    attn_kernel<<<agrid, dim3(128), 0, stream>>>(qb, kb, vb, ab);

    // Output projection (bf16 A via async DMA, fp32 W -> fp32 out)
    gemm_bias_kernel<true, true><<<ggrid, gblk, 0, stream>>>(ab, wo_w, wo_b, (float*)d_out, MTOT, DIM, DIM);
}
